// GroupedQueryAttention_21414706937867
// MI455X (gfx1250) — compile-verified, matrix operand order checked by reading, not run on hardware
//
#include <hip/hip_runtime.h>
#include <hip/hip_bf16.h>

// Problem constants (reference: B,T,C=2,2048,2048 NH=32 NKV=8 HD=64 G=4)
#define B_   2
#define T_   2048
#define C_   2048
#define NH_  32
#define NKV_ 8
#define HD_  64
#define G_   4
#define BT_  (B_ * T_)        // 4096
#define KVC_ (NKV_ * HD_)     // 512

typedef __bf16 bf16_t;
typedef __attribute__((ext_vector_type(8)))  bf16_t v8bf;
typedef __attribute__((ext_vector_type(16))) bf16_t v16bf;
typedef __attribute__((ext_vector_type(8)))  float  v8f;
typedef __attribute__((ext_vector_type(4)))  unsigned int v4u;
typedef __attribute__((ext_vector_type(8)))  int v8i;
typedef __attribute__((ext_vector_type(4)))  int v4i;

union Frag {
    v16bf  v;
    v8bf   h[2];
    bf16_t e[16];
};

__device__ __forceinline__ v8f wmma_bf16(v16bf a, v16bf b, v8f c) {
    // 8 args: (neg_a, A, neg_b, B, c_mod, C, reuse_a, reuse_b)
    return __builtin_amdgcn_wmma_f32_16x16x32_bf16(false, a, false, b, (short)0, c,
                                                   false, false);
}

// TDM: DMA a 2D bf16 tile (tile0 contiguous elems x tile1 rows, row stride
// stride0 elems) from global into LDS at byte offset ldsOff. Wave-level op,
// tracked with TENSORcnt. D# packed per CDNA5 ISA 8.3/8.4.
__device__ __forceinline__ void tdm_load_2d(unsigned ldsOff, const void* gptr,
                                            unsigned tdim0, unsigned tdim1,
                                            unsigned tile0, unsigned tile1,
                                            unsigned stride0)
{
    const size_t ga = (size_t)gptr;
    v4u g0 = { 1u,                                   // count=1, user D#
               ldsOff,                               // lds_addr
               (unsigned)(ga & 0xffffffffu),         // global_addr lo
               (unsigned)((ga >> 32) & 0x01ffffffu)  // global_addr hi
                   | 0x80000000u };                  // type=2 (image)
    v8i g1 = { (int)0x00010000,                      // data_size = 2 bytes
               (int)((tdim0 & 0xffffu) << 16),       // tensor_dim0 lo16
               (int)(((tdim0 >> 16) & 0xffffu) | ((tdim1 & 0xffffu) << 16)),
               (int)(((tdim1 >> 16) & 0xffffu) | ((tile0 & 0xffffu) << 16)),
               (int)(tile1 & 0xffffu),               // tile_dim1 (tile_dim2=0)
               (int)stride0,                         // tensor_dim0_stride lo32
               0, 0 };
    v4i gz = { 0, 0, 0, 0 };
#if __clang_major__ >= 23
    v8i gz8 = { 0, 0, 0, 0, 0, 0, 0, 0 };
    __builtin_amdgcn_tensor_load_to_lds(g0, g1, gz, gz, gz8, 0);
#else
    __builtin_amdgcn_tensor_load_to_lds(g0, g1, gz, gz, 0);
#endif
}

// ---------------------------------------------------------------- converts
__global__ __launch_bounds__(256) void k_cvt_bf16(const float* __restrict__ s,
                                                  bf16_t* __restrict__ d, long n) {
    long i = (long)blockIdx.x * blockDim.x + threadIdx.x;
    if (i < n) d[i] = (bf16_t)s[i];
}

// src: K x N row-major (f32)  ->  dst: N x K row-major (bf16)
__global__ __launch_bounds__(256) void k_cvt_transpose(const float* __restrict__ s,
                                                       bf16_t* __restrict__ d,
                                                       int K, int N) {
    long i = (long)blockIdx.x * blockDim.x + threadIdx.x;
    if (i >= (long)K * N) return;
    int k = (int)(i / N);
    int n = (int)(i - (long)k * N);
    d[(long)n * K + k] = (bf16_t)s[i];
}

// ---------------------------------------------------------------- GEMM
// C(MxN) = A(MxK bf16 row-major) * Bt(NxK bf16 row-major)^T
// Block = 8 waves (2x4) -> 128x256 macro-tile; wave -> 64x64 (4x4 WMMA accs).
// K stepped by 32 with TDM-filled, double-buffered LDS slabs:
//   sA: 128x32 (8 KB), sB: 256x32 (16 KB), x2 buffers = 48 KB.
// Every wave DMAs its own 1/8 slice each step (TDM ignores EXEC, so issue
// sites are never lane-divergent), computes from the current buffer, then
// s_wait_tensorcnt + barrier. DMA of step k+1 overlaps the 16 WMMAs of step k.
template <int OUTF32>
__global__ __launch_bounds__(256) void k_gemm_wmma(
        const bf16_t* __restrict__ A, int lda,
        const bf16_t* __restrict__ Bt, int ldb,
        bf16_t* __restrict__ Cb, float* __restrict__ Cf, int ldc,
        const float* __restrict__ bias,
        int M, int N, int K)
{
    __shared__ bf16_t sA[2][128 * 32];
    __shared__ bf16_t sB[2][256 * 32];

    const int lane = threadIdx.x & 31;
    const int wv   = threadIdx.x >> 5;
    const int wm   = wv >> 2;            // 0..1 in M
    const int wn   = wv & 3;             // 0..3 in N
    const int tilesN = N >> 8;           // macro tiles of 256
    const int mb = blockIdx.x / tilesN;
    const int nb = blockIdx.x - mb * tilesN;
    const int r  = lane & 15;
    const int hf = lane >> 4;

    const unsigned offA[2] = { (unsigned)(size_t)(void*)&sA[0][0],
                               (unsigned)(size_t)(void*)&sA[1][0] };
    const unsigned offB[2] = { (unsigned)(size_t)(void*)&sB[0][0],
                               (unsigned)(size_t)(void*)&sB[1][0] };

    const int aRow0 = mb * 128 + wv * 16;   // this wave's A slice: 16 rows
    const int bRow0 = nb * 256 + wv * 32;   // this wave's B slice: 32 rows

    v8f acc[4][4];
#pragma unroll
    for (int i = 0; i < 4; ++i)
#pragma unroll
        for (int j = 0; j < 4; ++j)
#pragma unroll
            for (int q = 0; q < 8; ++q) acc[i][j][q] = 0.0f;

    // preload k-slice 0
    tdm_load_2d(offA[0] + wv * 16 * 64, A + (size_t)aRow0 * lda,
                K, M - aRow0, 32, 16, lda);
    tdm_load_2d(offB[0] + wv * 32 * 64, Bt + (size_t)bRow0 * ldb,
                K, N - bRow0, 32, 32, ldb);
    __builtin_amdgcn_s_wait_tensorcnt(0);
    __syncthreads();

    int buf = 0;
    for (int kk = 0; kk < K; kk += 32) {
        const int nxt = buf ^ 1;
        if (kk + 32 < K) {   // uniform branch: every wave issues its DMA slice
            tdm_load_2d(offA[nxt] + wv * 16 * 64,
                        A + (size_t)aRow0 * lda + kk + 32,
                        K - kk - 32, M - aRow0, 32, 16, lda);
            tdm_load_2d(offB[nxt] + wv * 32 * 64,
                        Bt + (size_t)bRow0 * ldb + kk + 32,
                        K - kk - 32, N - bRow0, 32, 32, ldb);
        }

        const bf16_t* sa = &sA[buf][0];
        const bf16_t* sb = &sB[buf][0];
        Frag af[4], bq[4];
#pragma unroll
        for (int i = 0; i < 4; ++i) {
            const bf16_t* p = sa + (wm * 64 + i * 16 + r) * 32 + hf * 8;
            af[i].h[0] = *(const v8bf*)(p);
            af[i].h[1] = *(const v8bf*)(p + 16);
        }
#pragma unroll
        for (int j = 0; j < 4; ++j) {
            const bf16_t* p = sb + (wn * 64 + j * 16 + r) * 32 + hf * 8;
            bq[j].h[0] = *(const v8bf*)(p);
            bq[j].h[1] = *(const v8bf*)(p + 16);
        }
#pragma unroll
        for (int i = 0; i < 4; ++i)
#pragma unroll
            for (int j = 0; j < 4; ++j)
                acc[i][j] = wmma_bf16(af[i].v, bq[j].v, acc[i][j]);

        __builtin_amdgcn_s_wait_tensorcnt(0);
        __syncthreads();
        buf = nxt;
    }

    // C/D layout: lane holds column n = lane&15, rows q + 8*(lane>>4) in vgpr q
#pragma unroll
    for (int i = 0; i < 4; ++i) {
#pragma unroll
        for (int j = 0; j < 4; ++j) {
            const int col = nb * 256 + wn * 64 + j * 16 + r;
#pragma unroll
            for (int q = 0; q < 8; ++q) {
                const int row = mb * 128 + wm * 64 + i * 16 + hf * 8 + q;
                const float vv = acc[i][j][q];
                if (OUTF32) Cf[(size_t)row * ldc + col] = vv + bias[col];
                else        Cb[(size_t)row * ldc + col] = (bf16_t)vv;
            }
        }
    }
}

// ---------------------------------------------------------------- attention
// One wave (single-wave workgroup) per (b, head, 16-row q-tile).
// Per 32-key chunk: TDM DMA of the V tile into LDS overlapped with 4 score
// WMMAs, online softmax (max via lane shuffles; row-sum via one P*ones WMMA),
// LDS transpose of P, then 4 PV WMMAs.
__global__ __launch_bounds__(32) void k_attn(const bf16_t* __restrict__ Q,
                                             const bf16_t* __restrict__ Kk,
                                             const bf16_t* __restrict__ Vv,
                                             bf16_t* __restrict__ Y)
{
    __shared__ bf16_t ldsP[16 * 32];   // P tile, row-major 16x32
    __shared__ bf16_t ldsV[32 * 64];   // V chunk, row-major 32 keys x 64 d

    const int lane = threadIdx.x;
    const int n  = lane & 15;
    const int hf = lane >> 4;

    int idx = blockIdx.x;
    const int qt = idx & (T_ / 16 - 1); idx >>= 7;   // 128 q-tiles
    const int h  = idx & (NH_ - 1);     idx >>= 5;   // 32 heads
    const int b  = idx;
    const int hk = h >> 2;                           // h / G

    // Q fragments for d-chunks [0,32) and [32,64)
    Frag aq[2];
    {
        const bf16_t* qp =
            Q + ((size_t)(b * T_ + qt * 16 + n) * NH_ + h) * HD_ + hf * 8;
#pragma unroll
        for (int c = 0; c < 2; ++c) {
            aq[c].h[0] = *(const v8bf*)(qp + c * 32);
            aq[c].h[1] = *(const v8bf*)(qp + c * 32 + 16);
        }
    }

    Frag onesF;
#pragma unroll
    for (int j = 0; j < 16; ++j) onesF.e[j] = (bf16_t)1.0f;

    float mrow[8], lrow[8], corrA[8];
    v8f accO[4];
#pragma unroll
    for (int q = 0; q < 8; ++q) { mrow[q] = -1e30f; lrow[q] = 0.0f; }
#pragma unroll
    for (int dt = 0; dt < 4; ++dt)
#pragma unroll
        for (int q = 0; q < 8; ++q) accO[dt][q] = 0.0f;

    const unsigned ldsVoff = (unsigned)(size_t)(void*)ldsV;

    const int nch = (qt * 16 + 47) >> 5;   // ceil((qt*16+16)/32)
    for (int ch = 0; ch < nch; ++ch) {
        const int c32 = ch * 32;

        // ---- TDM: DMA 32x64 bf16 V tile into LDS. tensor_dim1 = remaining
        // rows so the tail chunk zero-fills instead of reading OOB (zero V
        // rows pair with P == 0 from masked scores).
        {
            // TDM is unordered vs DS ops: prior-chunk LDS reads must be done
            asm volatile("s_wait_dscnt 0x0" ::: "memory");
            tdm_load_2d(ldsVoff,
                        Vv + ((size_t)(b * T_ + c32) * NKV_ + hk) * HD_,
                        HD_, T_ - c32, HD_, 32, KVC_);
        }

        // ---- scores S = Q * K^T for two 16-key tiles (overlaps the DMA)
        v8f s0, s1;
#pragma unroll
        for (int q = 0; q < 8; ++q) { s0[q] = 0.0f; s1[q] = 0.0f; }
        {
            int k0 = c32 + n;      if (k0 > T_ - 1) k0 = T_ - 1;
            int k1 = c32 + 16 + n; if (k1 > T_ - 1) k1 = T_ - 1;
            const bf16_t* kp0 =
                Kk + ((size_t)(b * T_ + k0) * NKV_ + hk) * HD_ + hf * 8;
            const bf16_t* kp1 =
                Kk + ((size_t)(b * T_ + k1) * NKV_ + hk) * HD_ + hf * 8;
            Frag bk;
#pragma unroll
            for (int c = 0; c < 2; ++c) {
                bk.h[0] = *(const v8bf*)(kp0 + c * 32);
                bk.h[1] = *(const v8bf*)(kp0 + c * 32 + 16);
                s0 = wmma_bf16(aq[c].v, bk.v, s0);
            }
#pragma unroll
            for (int c = 0; c < 2; ++c) {
                bk.h[0] = *(const v8bf*)(kp1 + c * 32);
                bk.h[1] = *(const v8bf*)(kp1 + c * 32 + 16);
                s1 = wmma_bf16(aq[c].v, bk.v, s1);
            }
        }

        // ---- online softmax: max via 16-lane-half shuffle reductions
#pragma unroll
        for (int vr = 0; vr < 8; ++vr) {
            const int qrow = qt * 16 + vr + 8 * hf;
            float a0 = (c32 + n      <= qrow) ? s0[vr] * 0.125f : -1e30f;
            float a1 = (c32 + 16 + n <= qrow) ? s1[vr] * 0.125f : -1e30f;
            float mx = fmaxf(a0, a1);
#pragma unroll
            for (int off = 8; off; off >>= 1)
                mx = fmaxf(mx, __shfl_xor(mx, off, 32));
            const float mnew = fmaxf(mrow[vr], mx);
            const float corr = __expf(mrow[vr] - mnew);
            const float p0 = __expf(a0 - mnew);
            const float p1 = __expf(a1 - mnew);
            corrA[vr] = corr;
            mrow[vr] = mnew;
#pragma unroll
            for (int dt = 0; dt < 4; ++dt) accO[dt][vr] *= corr;
            ldsP[(vr + 8 * hf) * 32 + n]      = (bf16_t)p0;
            ldsP[(vr + 8 * hf) * 32 + 16 + n] = (bf16_t)p1;
        }
        __syncthreads();   // single-wave WG: ordering only

        // ---- row sums on the matrix pipe: l += rowsum(P) = P x ones
        Frag pa;
        pa.h[0] = *(const v8bf*)&ldsP[n * 32 + hf * 8];
        pa.h[1] = *(const v8bf*)&ldsP[n * 32 + hf * 8 + 16];
        {
            v8f zz;
#pragma unroll
            for (int q = 0; q < 8; ++q) zz[q] = 0.0f;
            v8f sumT = wmma_bf16(pa.v, onesF.v, zz);
#pragma unroll
            for (int vr = 0; vr < 8; ++vr)
                lrow[vr] = lrow[vr] * corrA[vr] + sumT[vr];
        }

        // ---- O += P(16x32) x V(32x64); V comes from the TDM-filled LDS
        __builtin_amdgcn_s_wait_tensorcnt(0);
#pragma unroll
        for (int dt = 0; dt < 4; ++dt) {
            Frag bv;
#pragma unroll
            for (int j = 0; j < 16; ++j) {
                const int kkey = (j < 8) ? (hf * 8 + j) : (16 + hf * 8 + (j - 8));
                bv.e[j] = ldsV[kkey * 64 + dt * 16 + n];
            }
            accO[dt] = wmma_bf16(pa.v, bv.v, accO[dt]);
        }
        __syncthreads();
    }

    // normalize and write Y (B,T,NH,HD) bf16
#pragma unroll
    for (int dt = 0; dt < 4; ++dt)
#pragma unroll
        for (int vr = 0; vr < 8; ++vr) {
            const int qrow = qt * 16 + vr + 8 * hf;
            const float ov = accO[dt][vr] / lrow[vr];
            Y[((size_t)(b * T_ + qrow) * NH_ + h) * HD_ + dt * 16 + n] = (bf16_t)ov;
        }
}

// ---------------------------------------------------------------- launch
extern "C" void kernel_launch(void* const* d_in, const int* in_sizes, int n_in,
                              void* d_out, int out_size, void* d_ws, size_t ws_size,
                              hipStream_t stream)
{
    const float* x  = (const float*)d_in[0];
    const float* wq = (const float*)d_in[1];
    const float* wk = (const float*)d_in[2];
    const float* wv = (const float*)d_in[3];
    const float* wp = (const float*)d_in[4];
    const float* bp = (const float*)d_in[5];
    float* out = (float*)d_out;

    char* ws = (char*)d_ws;
    size_t off = 0;
    auto walloc = [&](size_t bytes) -> char* {
        char* p = ws + off;
        off += (bytes + 255) & ~(size_t)255;
        return p;
    };
    bf16_t* xb  = (bf16_t*)walloc((size_t)BT_ * C_   * 2);
    bf16_t* wqT = (bf16_t*)walloc((size_t)C_  * C_   * 2);
    bf16_t* wkT = (bf16_t*)walloc((size_t)KVC_ * C_  * 2);
    bf16_t* wvT = (bf16_t*)walloc((size_t)KVC_ * C_  * 2);
    bf16_t* wpT = (bf16_t*)walloc((size_t)C_  * C_   * 2);
    bf16_t* Qb  = (bf16_t*)walloc((size_t)BT_ * C_   * 2);
    bf16_t* Kb  = (bf16_t*)walloc((size_t)BT_ * KVC_ * 2);
    bf16_t* Vb  = (bf16_t*)walloc((size_t)BT_ * KVC_ * 2);
    bf16_t* Yb  = (bf16_t*)walloc((size_t)BT_ * C_   * 2);

    {   // x -> bf16
        long nel = (long)BT_ * C_;
        k_cvt_bf16<<<(int)((nel + 255) / 256), 256, 0, stream>>>(x, xb, nel);
    }
    {   // weights -> bf16, transposed to N x K
        long nqq = (long)C_ * C_;
        long nkv = (long)C_ * KVC_;
        k_cvt_transpose<<<(int)((nqq + 255) / 256), 256, 0, stream>>>(wq, wqT, C_, C_);
        k_cvt_transpose<<<(int)((nkv + 255) / 256), 256, 0, stream>>>(wk, wkT, C_, KVC_);
        k_cvt_transpose<<<(int)((nkv + 255) / 256), 256, 0, stream>>>(wv, wvT, C_, KVC_);
        k_cvt_transpose<<<(int)((nqq + 255) / 256), 256, 0, stream>>>(wp, wpT, C_, C_);
    }

    // projections: Q (4096x2048x2048), K/V (4096x512x2048)
    {
        int blocks = (BT_ / 128) * (C_ / 256);       // 32*8 = 256
        k_gemm_wmma<0><<<blocks, 256, 0, stream>>>(
            xb, C_, wqT, C_, Qb, nullptr, C_, nullptr, BT_, C_, C_);
    }
    {
        int blocks = (BT_ / 128) * (KVC_ / 256);     // 32*2 = 64
        k_gemm_wmma<0><<<blocks, 256, 0, stream>>>(
            xb, C_, wkT, C_, Kb, nullptr, KVC_, nullptr, BT_, KVC_, C_);
        k_gemm_wmma<0><<<blocks, 256, 0, stream>>>(
            xb, C_, wvT, C_, Vb, nullptr, KVC_, nullptr, BT_, KVC_, C_);
    }

    // causal GQA flash attention
    k_attn<<<B_ * NH_ * (T_ / 16), 32, 0, stream>>>(Qb, Kb, Vb, Yb);

    // output projection + bias, f32 out
    {
        int blocks = (BT_ / 128) * (C_ / 256);
        k_gemm_wmma<1><<<blocks, 256, 0, stream>>>(
            Yb, C_, wpT, C_, nullptr, out, C_, bp, BT_, C_, C_);
    }
}
